// EdgeConv_6150393168310
// MI455X (gfx1250) — matrix operand order repaired and checked by reading, not compiled, not run on hardware
//
#include <hip/hip_runtime.h>

typedef __attribute__((ext_vector_type(16))) _Float16 v16h;
typedef __attribute__((ext_vector_type(8)))  _Float16 v8h;
typedef __attribute__((ext_vector_type(8)))  float    v8f;
typedef __attribute__((ext_vector_type(4)))  unsigned int u32x4;
typedef __attribute__((ext_vector_type(8)))  int          i32x8;
typedef __attribute__((ext_vector_type(4)))  int          i32x4;

#define B_    8
#define CIN   64
#define NPTS  4096
#define KNN   20
#define COUT  64
#define EPSV  1e-5f

static __device__ __forceinline__ v16h cat8(v8h lo, v8h hi) {
    return __builtin_shufflevector(lo, hi, 0,1,2,3,4,5,6,7,8,9,10,11,12,13,14,15);
}
static __device__ __forceinline__ v8f wmma16(v16h a, v16h b, v8f c) {
    // emits v_wmma_f32_16x16x32_f16
    return __builtin_amdgcn_wmma_f32_16x16x32_f16(false, a, false, b, (short)0, c, false, false);
}

// Issue one TDM descriptor: 1-D copy of `nelem` f32 from global -> LDS.
// D# layout per CDNA5 ISA ch.8 (group0: count/lds_addr/global_addr/type=2,
// group1: data_size=4B, tensor_dim0 = tile_dim0 = nelem).
// This toolchain exposes the 6-arg builtin: (g0, g1, g2, g3, g?, cpol).
static __device__ __forceinline__ void tdm_copy_f32(const float* gsrc, void* lds_dst,
                                                    unsigned nelem) {
    unsigned long long ga = (unsigned long long)(size_t)gsrc;
    unsigned lds = (unsigned)(size_t)lds_dst;         // low 32 bits = LDS byte offset
    u32x4 g0;
    g0.x = 1u;                                        // count = 1 valid descriptor
    g0.y = lds;                                       // lds_addr
    g0.z = (unsigned)ga;                              // global_addr[31:0]
    g0.w = (unsigned)(ga >> 32) | (2u << 30);         // global_addr[56:32] | type=2
    i32x8 g1;
    g1[0] = (int)(2u << 16);                          // data_size = 2 (4 bytes)
    g1[1] = (int)((nelem & 0xFFFFu) << 16);           // tensor_dim0[15:0]
    g1[2] = (int)(((nelem >> 16) & 0xFFFFu) | (1u << 16)); // tensor_dim0[31:16] | tensor_dim1=1
    g1[3] = (int)((nelem & 0xFFFFu) << 16);           // tile_dim0
    g1[4] = 0;                                        // tile_dim1/2 unused
    g1[5] = (int)nelem;                               // tensor_dim0_stride[31:0]
    g1[6] = 0;
    g1[7] = 0;
    i32x4 z4 = {0, 0, 0, 0};
    i32x8 z8 = {0, 0, 0, 0, 0, 0, 0, 0};
    __builtin_amdgcn_tensor_load_to_lds(g0, g1, z4, z4, z8, 0);
}

// ---------------------------------------------------------------- sq norms
__global__ void sqnorm_kernel(const float* __restrict__ x, float* __restrict__ xx) {
    int gid = blockIdx.x * blockDim.x + threadIdx.x;     // over B*N
    int b = gid >> 12, n = gid & (NPTS - 1);
    const float* xb = x + (size_t)b * CIN * NPTS + n;
    float s = 0.f;
    #pragma unroll
    for (int c = 0; c < CIN; ++c) { float v = xb[(size_t)c * NPTS]; s += v * v; }
    xx[gid] = s;
}

// ---------------------------------------------------------------- kNN (WMMA Gram + fused top-k)
__global__ void __launch_bounds__(32) knn_kernel(const float* __restrict__ x,
                                                 const float* __restrict__ xx,
                                                 int* __restrict__ idx_out) {
    __shared__ float     xx_s[NPTS];          // 16 KB (filled by TDM)
    __shared__ _Float16  Al[32][64];          // 4 KB   A rows (k contiguous)
    __shared__ _Float16  Bl[16][64];          // 2 KB   [col][k]
    __shared__ float     Dl[32][17];          // 2.2 KB inner-product tile

    const int tid  = threadIdx.x;
    const int wg   = blockIdx.x;
    const int b    = wg >> 7;                  // 128 blocks per batch
    const int n0   = (wg & 127) << 5;          // 32 rows per block
    const float* xb  = x  + (size_t)b * CIN * NPTS;
    const float* xxb = xx + (size_t)b * NPTS;

    // async DMA of the squared-norm row into LDS (overlaps with Al staging below)
    tdm_copy_f32(xxb, xx_s, NPTS);

    for (int e = tid; e < 32 * 64; e += 32) {
        int nl = e & 31, k = e >> 5;
        Al[nl][k] = (_Float16)xb[(size_t)k * NPTS + n0 + nl];
    }

    float bd[KNN]; int bi[KNN];
    #pragma unroll
    for (int q = 0; q < KNN; ++q) { bd[q] = -3.0e38f; bi[q] = 0; }

    const int hlf = tid >> 4, col = tid & 15;
    const float xxr = xxb[n0 + tid];

    __builtin_amdgcn_s_wait_tensorcnt(0);      // xx_s resident
    __syncthreads();

    for (int m0 = 0; m0 < NPTS; m0 += 16) {
        // prefetch next column tile's source rows (2 k-rows per lane covers all 64)
        if (m0 + 16 < NPTS) {
            __builtin_prefetch(&xb[(size_t)(tid * 2) * NPTS + m0 + 16], 0, 0);
            __builtin_prefetch(&xb[(size_t)(tid * 2 + 1) * NPTS + m0 + 16], 0, 0);
        }
        for (int e = tid; e < 16 * 64; e += 32) {
            int cc = e & 15, k = e >> 4;
            Bl[cc][k] = (_Float16)xb[(size_t)k * NPTS + m0 + cc];
        }
        __syncthreads();

        v8f acc0 = {}, acc1 = {};
        #pragma unroll
        for (int kb = 0; kb < 64; kb += 32) {
            v8h blo = *(const v8h*)&Bl[col][kb + 8 * hlf];
            v8h bhi = *(const v8h*)&Bl[col][kb + 16 + 8 * hlf];
            v16h bf = cat8(blo, bhi);
            v8h a0l = *(const v8h*)&Al[col][kb + 8 * hlf];
            v8h a0h = *(const v8h*)&Al[col][kb + 16 + 8 * hlf];
            v8h a1l = *(const v8h*)&Al[16 + col][kb + 8 * hlf];
            v8h a1h = *(const v8h*)&Al[16 + col][kb + 16 + 8 * hlf];
            acc0 = wmma16(cat8(a0l, a0h), bf, acc0);
            acc1 = wmma16(cat8(a1l, a1h), bf, acc1);
        }
        #pragma unroll
        for (int v = 0; v < 8; ++v) {
            Dl[v + 8 * hlf][col]      = acc0[v];
            Dl[16 + v + 8 * hlf][col] = acc1[v];
        }
        __syncthreads();

        // lane tid owns row n0+tid; scan 16 candidate columns
        #pragma unroll
        for (int j = 0; j < 16; ++j) {
            float d = 2.0f * Dl[tid][j] - xxr - xx_s[m0 + j];
            if (d > bd[KNN - 1]) {
                bd[KNN - 1] = d; bi[KNN - 1] = m0 + j;
                #pragma unroll
                for (int q = KNN - 1; q > 0; --q) {
                    if (bd[q] > bd[q - 1]) {
                        float td = bd[q]; bd[q] = bd[q - 1]; bd[q - 1] = td;
                        int   ti = bi[q]; bi[q] = bi[q - 1]; bi[q - 1] = ti;
                    }
                }
            }
        }
        __syncthreads();
    }

    int* op = idx_out + ((size_t)b * NPTS + n0 + tid) * KNN;
    #pragma unroll
    for (int q = 0; q < KNN; ++q) op[q] = bi[q];
}

// ---------------------------------------------------------------- edge GEMM pass 1: BN partials
__global__ void __launch_bounds__(32) edge_stats_kernel(const float* __restrict__ x,
                                                        const int* __restrict__ idx,
                                                        const float* __restrict__ W,
                                                        float* __restrict__ part,
                                                        int ngroups) {
    __shared__ _Float16 Wl[64][128];   // 16 KB  W[o][c], c contiguous
    __shared__ float    ctr[4][64];    // 1 KB
    __shared__ int      jbuf[80];
    __shared__ _Float16 el[16][128];   // 4 KB edge rows

    const int tid = threadIdx.x;
    const int hlf = tid >> 4, col = tid & 15;
    for (int e = tid; e < 64 * 128; e += 32) {
        int c = e & 127, o = e >> 7;
        Wl[o][c] = (_Float16)W[o * 128 + c];
    }

    float ls[4] = {0.f, 0.f, 0.f, 0.f}, lq[4] = {0.f, 0.f, 0.f, 0.f};

    for (int g = blockIdx.x; g < ngroups; g += gridDim.x) {
        int b  = g >> 10;                 // 1024 groups per batch
        int nb = (g & 1023) << 2;         // 4 points per group
        const float* xb = x + (size_t)b * CIN * NPTS;
        __syncthreads();
        for (int e = tid; e < 256; e += 32) {
            int nl = e & 3, c = e >> 2;
            ctr[nl][c] = xb[(size_t)c * NPTS + nb + nl];
        }
        for (int e = tid; e < 80; e += 32)
            jbuf[e] = idx[((size_t)b * NPTS + nb + e / 20) * KNN + e % 20];
        __syncthreads();

        for (int tt = 0; tt < 5; ++tt) {              // 5 x 16 rows = 4*20
            int rloc = tt * 16 + col;
            int nl = rloc / 20;
            int j  = jbuf[rloc];
            if (hlf == 0) {
                #pragma unroll
                for (int c = 0; c < 64; ++c) el[col][c] = (_Float16)ctr[nl][c];
            } else {
                for (int c = 0; c < 64; ++c)
                    el[col][64 + c] = (_Float16)(xb[(size_t)c * NPTS + j] - ctr[nl][c]);
            }
            __syncthreads();

            v8f acc[4] = {};
            #pragma unroll
            for (int kb = 0; kb < 128; kb += 32) {
                v8h alo = *(const v8h*)&el[col][kb + 8 * hlf];
                v8h ahi = *(const v8h*)&el[col][kb + 16 + 8 * hlf];
                v16h af = cat8(alo, ahi);
                #pragma unroll
                for (int t = 0; t < 4; ++t) {
                    v8h blo = *(const v8h*)&Wl[16 * t + col][kb + 8 * hlf];
                    v8h bhi = *(const v8h*)&Wl[16 * t + col][kb + 16 + 8 * hlf];
                    acc[t] = wmma16(af, cat8(blo, bhi), acc[t]);
                }
            }
            #pragma unroll
            for (int t = 0; t < 4; ++t)
                #pragma unroll
                for (int v = 0; v < 8; ++v) { float vv = acc[t][v]; ls[t] += vv; lq[t] += vv * vv; }
            __syncthreads();
        }
    }

    // deterministic block partials: channel = 16*t + col
    #pragma unroll
    for (int t = 0; t < 4; ++t) {
        float s = ls[t] + __shfl_xor(ls[t], 16, 32);
        float q = lq[t] + __shfl_xor(lq[t], 16, 32);
        if (hlf == 0) {
            part[(size_t)blockIdx.x * 128 + 16 * t + col]      = s;
            part[(size_t)blockIdx.x * 128 + 64 + 16 * t + col] = q;
        }
    }
}

// ---------------------------------------------------------------- fold partials -> scale/bias
__global__ void finalize_kernel(const float* __restrict__ part,
                                const float* __restrict__ gamma,
                                const float* __restrict__ beta,
                                float* __restrict__ scale,
                                float* __restrict__ bias, int npart) {
    int o = threadIdx.x;
    if (o >= COUT) return;
    float s = 0.f, q = 0.f;
    for (int p = 0; p < npart; ++p) {
        s += part[(size_t)p * 128 + o];
        q += part[(size_t)p * 128 + 64 + o];
    }
    const float cnt = (float)B_ * (float)NPTS * (float)KNN;
    float mean = s / cnt;
    float var  = q / cnt - mean * mean;
    float sc   = gamma[o] * rsqrtf(var + EPSV);
    scale[o] = sc;
    bias[o]  = beta[o] - mean * sc;
}

// ---------------------------------------------------------------- edge GEMM pass 2: BN+LReLU+max
__global__ void __launch_bounds__(32) edge_out_kernel(const float* __restrict__ x,
                                                      const int* __restrict__ idx,
                                                      const float* __restrict__ W,
                                                      const float* __restrict__ scale,
                                                      const float* __restrict__ bias,
                                                      float* __restrict__ out) {
    __shared__ _Float16 Wl[64][128];
    __shared__ float    ctr[4][64];
    __shared__ int      jbuf[80];
    __shared__ _Float16 el[16][128];
    __shared__ float    hbuf[80][64];  // 20 KB normalized h

    const int tid = threadIdx.x;
    const int hlf = tid >> 4, col = tid & 15;
    for (int e = tid; e < 64 * 128; e += 32) {
        int c = e & 127, o = e >> 7;
        Wl[o][c] = (_Float16)W[o * 128 + c];
    }
    float sc[4], bs[4];
    #pragma unroll
    for (int t = 0; t < 4; ++t) { sc[t] = scale[16 * t + col]; bs[t] = bias[16 * t + col]; }

    const int g  = blockIdx.x;
    const int b  = g >> 10;
    const int nb = (g & 1023) << 2;
    const float* xb = x + (size_t)b * CIN * NPTS;

    for (int e = tid; e < 256; e += 32) {
        int nl = e & 3, c = e >> 2;
        ctr[nl][c] = xb[(size_t)c * NPTS + nb + nl];
    }
    for (int e = tid; e < 80; e += 32)
        jbuf[e] = idx[((size_t)b * NPTS + nb + e / 20) * KNN + e % 20];
    __syncthreads();

    for (int tt = 0; tt < 5; ++tt) {
        int rloc = tt * 16 + col;
        int nl = rloc / 20;
        int j  = jbuf[rloc];
        if (hlf == 0) {
            #pragma unroll
            for (int c = 0; c < 64; ++c) el[col][c] = (_Float16)ctr[nl][c];
        } else {
            for (int c = 0; c < 64; ++c)
                el[col][64 + c] = (_Float16)(xb[(size_t)c * NPTS + j] - ctr[nl][c]);
        }
        __syncthreads();

        v8f acc[4] = {};
        #pragma unroll
        for (int kb = 0; kb < 128; kb += 32) {
            v8h alo = *(const v8h*)&el[col][kb + 8 * hlf];
            v8h ahi = *(const v8h*)&el[col][kb + 16 + 8 * hlf];
            v16h af = cat8(alo, ahi);
            #pragma unroll
            for (int t = 0; t < 4; ++t) {
                v8h blo = *(const v8h*)&Wl[16 * t + col][kb + 8 * hlf];
                v8h bhi = *(const v8h*)&Wl[16 * t + col][kb + 16 + 8 * hlf];
                acc[t] = wmma16(af, cat8(blo, bhi), acc[t]);
            }
        }
        #pragma unroll
        for (int t = 0; t < 4; ++t)
            #pragma unroll
            for (int v = 0; v < 8; ++v) {
                float vv = acc[t][v] * sc[t] + bs[t];
                vv = vv >= 0.f ? vv : 0.2f * vv;                  // LeakyReLU(0.2)
                hbuf[tt * 16 + v + 8 * hlf][16 * t + col] = vv;
            }
        __syncthreads();
    }

    // max over the 20 neighbours; 4 points x 64 channels = 256 outputs
    for (int p = tid; p < 256; p += 32) {
        int nl = p >> 6, ch = p & 63;
        float m = -3.0e38f;
        #pragma unroll
        for (int kk = 0; kk < KNN; ++kk) m = fmaxf(m, hbuf[nl * 20 + kk][ch]);
        out[((size_t)b * COUT + ch) * NPTS + nb + nl] = m;
    }
}

// ---------------------------------------------------------------- launcher
extern "C" void kernel_launch(void* const* d_in, const int* in_sizes, int n_in,
                              void* d_out, int out_size, void* d_ws, size_t ws_size,
                              hipStream_t stream) {
    (void)in_sizes; (void)n_in; (void)out_size; (void)ws_size;
    const float* x     = (const float*)d_in[0];
    const float* W     = (const float*)d_in[1];
    const float* gamma = (const float*)d_in[2];
    const float* beta  = (const float*)d_in[3];
    float* out = (float*)d_out;

    char* ws = (char*)d_ws;
    float* xx    = (float*)ws;                                     // 32768 f = 128 KB
    int*   idx   = (int*)(ws + (size_t)32768 * 4);                 // 655360 i = 2.5 MB
    float* scale = (float*)(ws + (size_t)32768 * 4 + (size_t)655360 * 4);
    float* bias  = scale + 64;
    float* part  = bias + 64;                                      // 1024 * 128 f = 512 KB

    const int STAT_BLOCKS = 1024;
    const int NGROUPS     = (B_ * NPTS) / 4;                       // 8192

    sqnorm_kernel    <<<(B_ * NPTS) / 256, 256, 0, stream>>>(x, xx);
    knn_kernel       <<<(B_ * NPTS) / 32,   32, 0, stream>>>(x, xx, idx);
    edge_stats_kernel<<<STAT_BLOCKS,        32, 0, stream>>>(x, idx, W, part, NGROUPS);
    finalize_kernel  <<<1,                  64, 0, stream>>>(part, gamma, beta, scale, bias, STAT_BLOCKS);
    edge_out_kernel  <<<NGROUPS,            32, 0, stream>>>(x, idx, W, scale, bias, out);
}
